// MultiHeadAttention_4801773437478
// MI455X (gfx1250) — compile-verified, matrix operand order checked by reading, not run on hardware
//
#include <hip/hip_runtime.h>
#include <hip/hip_bf16.h>

typedef __attribute__((ext_vector_type(16))) _Float16 v16h;
typedef __attribute__((ext_vector_type(4)))  _Float16 v4h;
typedef __attribute__((ext_vector_type(8)))  float    v8f;

#define B_SZ   4
#define T_SZ   2048
#define E_SZ   1024
#define H_SZ   16
#define DK_SZ  64
#define NEG_INF (-1e30f)

__device__ __forceinline__ int lane_id() { return threadIdx.x & 31; }
__device__ __forceinline__ int wave_id() { return threadIdx.x >> 5; }

// D = A*B + C  (16x16x32, f16 in, f32 accum)
__device__ __forceinline__ v8f wmma16(v16h a, v16h b, v8f c) {
  return __builtin_amdgcn_wmma_f32_16x16x32_f16(false, a, false, b, (short)0, c,
                                                false, false);
}

// ---- CDNA5 async global->LDS copy (16B per lane), tracked by ASYNCcnt ----
__device__ __forceinline__ void async_copy_b128(void* lds_dst, const void* gsrc) {
  // flat address of a __shared__ object: LDS byte offset is addr[31:0]
  unsigned lds_addr = (unsigned)(unsigned long long)lds_dst;
  asm volatile("global_load_async_to_lds_b128 %0, %1, off"
               :: "v"(lds_addr), "v"(gsrc)
               : "memory");
}

__device__ __forceinline__ void wait_async0() {
#if __has_builtin(__builtin_amdgcn_s_wait_asynccnt)
  __builtin_amdgcn_s_wait_asynccnt(0);
#else
  asm volatile("s_wait_asynccnt 0x0" ::: "memory");
#endif
}

// CDNA5 16-bit A/B fragment K mapping within a 16x32 tile:
// k(vg, half, s) = (vg/4)*16 + half*8 + (vg%4)*2 + s
__device__ __forceinline__ v16h load_frag_rowmajor(const _Float16* __restrict__ src,
                                                   int ld) {
  const int l = lane_id();
  const int m = l & 15;
  const int hh = l >> 4;
  v16h f;
#pragma unroll
  for (int vg = 0; vg < 8; ++vg) {
    const int k0 = ((vg >> 2) << 4) + (hh << 3) + ((vg & 3) << 1);
    f[2 * vg]     = src[m * ld + k0];
    f[2 * vg + 1] = src[m * ld + k0 + 1];
  }
  return f;
}

// B-frag from a K-major (row-major [k][n]) source, leading dim `ld`.
__device__ __forceinline__ v16h load_frag_kmajor(const _Float16* __restrict__ src,
                                                 int ld) {
  const int l = lane_id();
  const int n = l & 15;
  const int hh = l >> 4;
  v16h f;
#pragma unroll
  for (int vg = 0; vg < 8; ++vg) {
    const int k0 = ((vg >> 2) << 4) + (hh << 3) + ((vg & 3) << 1);
    f[2 * vg]     = src[k0 * ld + n];
    f[2 * vg + 1] = src[(k0 + 1) * ld + n];
  }
  return f;
}

// ---------------------------------------------------------------- convert
__global__ __launch_bounds__(256) void cvt_f32_f16_kernel(
    const float* __restrict__ in, _Float16* __restrict__ out, int n4) {
  int i = blockIdx.x * 256 + threadIdx.x;
  if (i < n4) {
    float4 v = ((const float4*)in)[i];
    v4h h = {(_Float16)v.x, (_Float16)v.y, (_Float16)v.z, (_Float16)v.w};
    *(v4h*)(out + (size_t)i * 4) = h;
  }
}

// ---------------------------------------------------------------- GEMM
// out[m, f] = sum_e A[m, e] * Bw[f, e] + bias[f]
// MODE 0: scatter f16 into qkvh [B, H, T, 192]   (m = b*T + t, f = h*192 + j)
// MODE 1: write f32 [M, N] to out_f
template <int MODE>
__global__ __launch_bounds__(256) void gemm_f16_kernel(
    const _Float16* __restrict__ A,   // [M, K] row-major
    const _Float16* __restrict__ Bw,  // [N, K] row-major
    const float* __restrict__ bias,   // [N]
    _Float16* __restrict__ out_h, float* __restrict__ out_f,
    int M, int N, int K) {
  __shared__ __align__(16) _Float16 As[2][128][80];  // 160B rows, 16B aligned
  __shared__ __align__(16) _Float16 Bs[2][128][80];

  const int m0 = blockIdx.y * 128;
  const int n0 = blockIdx.x * 128;
  const int tid = threadIdx.x;
  const int wid = wave_id();
  const int waveM = wid >> 2;   // 0..1 -> 64 rows each
  const int waveN = wid & 3;    // 0..3 -> 32 cols each

  // async-stage one 128x64-half slice of A and B into buffer `buf`
  auto stage = [&](int buf, int k0) {
#pragma unroll
    for (int i = 0; i < 4; ++i) {
      const int idx = tid + i * 256;       // 0..1023
      const int row = idx >> 3;            // 0..127
      const int q = idx & 7;               // 8 x 16B chunks per row
      async_copy_b128(&As[buf][row][q * 8], &A[(size_t)(m0 + row) * K + k0 + q * 8]);
      async_copy_b128(&Bs[buf][row][q * 8], &Bw[(size_t)(n0 + row) * K + k0 + q * 8]);
    }
  };

  v8f acc[4][2];
#pragma unroll
  for (int i = 0; i < 4; ++i)
#pragma unroll
    for (int j = 0; j < 2; ++j) acc[i][j] = (v8f){0.f};

  const int nsteps = K / 64;
  stage(0, 0);
  wait_async0();
  __syncthreads();

  for (int s = 0; s < nsteps; ++s) {
    const int buf = s & 1;
    if (s + 1 < nsteps) stage(buf ^ 1, (s + 1) * 64);  // prefetch next tile

#pragma unroll
    for (int kk = 0; kk < 2; ++kk) {
      v16h af[4], bf[2];
#pragma unroll
      for (int i = 0; i < 4; ++i)
        af[i] = load_frag_rowmajor(&As[buf][waveM * 64 + i * 16][kk * 32], 80);
#pragma unroll
      for (int j = 0; j < 2; ++j)
        bf[j] = load_frag_rowmajor(&Bs[buf][waveN * 32 + j * 16][kk * 32], 80);
#pragma unroll
      for (int i = 0; i < 4; ++i)
#pragma unroll
        for (int j = 0; j < 2; ++j)
          acc[i][j] = wmma16(af[i], bf[j], acc[i][j]);
    }

    wait_async0();
    __syncthreads();
  }

  // epilogue
  const int l = lane_id();
  const int n_in = l & 15;
  const int hl = l >> 4;
#pragma unroll
  for (int i = 0; i < 4; ++i)
#pragma unroll
    for (int j = 0; j < 2; ++j)
#pragma unroll
      for (int r = 0; r < 8; ++r) {
        const int m = m0 + waveM * 64 + i * 16 + r + 8 * hl;
        const int n = n0 + waveN * 32 + j * 16 + n_in;
        const float val = acc[i][j][r] + bias[n];
        if (MODE == 0) {
          const int b = m / T_SZ, t = m % T_SZ;
          const int hh = n / 192, jj = n % 192;
          out_h[(((size_t)b * H_SZ + hh) * T_SZ + t) * 192 + jj] = (_Float16)val;
        } else {
          out_f[(size_t)m * N + n] = val;
        }
      }
}

// ---------------------------------------------------------------- attention
// qkvh: [B, H, T, 192] f16 (per row: q[0:64] k[64:128] v[128:192])
// Oh:   [B, T, E] f16
__global__ __launch_bounds__(256) void attn_kernel(
    const _Float16* __restrict__ qkvh, _Float16* __restrict__ Oh) {
  __shared__ __align__(16) _Float16 Ks[2][128][80];
  __shared__ __align__(16) _Float16 Vs[2][128][80];
  __shared__ __align__(16) _Float16 Ps[8][16][136];

  const int qb = blockIdx.x;            // query block (128 rows)
  const int bh = blockIdx.y;            // b*H + h
  const int b = bh >> 4, h = bh & 15;
  const _Float16* base = qkvh + (size_t)bh * T_SZ * 192;

  const int tid = threadIdx.x;
  const int wid = wave_id();            // wave owns 16 query rows
  const int l = lane_id();
  const int n_in = l & 15;
  const int hl = l >> 4;
  const int qrow0 = qb * 128 + wid * 16;
  const float scale = 0.125f;           // 1/sqrt(64)

  // async-stage K and V tiles of key block `kb` into buffer `buf`
  auto stageKV = [&](int buf, int kb) {
#pragma unroll
    for (int i = 0; i < 4; ++i) {
      const int idx = tid + i * 256;
      const int row = idx >> 3;
      const int q = idx & 7;
      const _Float16* kv = base + (size_t)(kb * 128 + row) * 192;
      async_copy_b128(&Ks[buf][row][q * 8], kv + 64 + q * 8);
      async_copy_b128(&Vs[buf][row][q * 8], kv + 128 + q * 8);
    }
  };

  // Q fragments (dk = 64 -> 2 k-steps), loaded directly from global
  v16h aq[2];
#pragma unroll
  for (int kk = 0; kk < 2; ++kk)
    aq[kk] = load_frag_rowmajor(base + (size_t)qrow0 * 192 + kk * 32, 192);

  v8f o[4];
#pragma unroll
  for (int j = 0; j < 4; ++j) o[j] = (v8f){0.f};
  float row_max[8], row_sum[8];
#pragma unroll
  for (int r = 0; r < 8; ++r) { row_max[r] = NEG_INF; row_sum[r] = 0.f; }

  stageKV(0, 0);
  wait_async0();
  __syncthreads();

  for (int kb = 0; kb <= qb; ++kb) {    // causal: only key blocks <= query block
    const int buf = kb & 1;
    if (kb + 1 <= qb) stageKV(buf ^ 1, kb + 1);  // prefetch next key block

    // S = Q * K^T  (16 x 128 per wave)
    v8f s[8];
#pragma unroll
    for (int nf = 0; nf < 8; ++nf) {
      v8f a = (v8f){0.f};
#pragma unroll
      for (int kk = 0; kk < 2; ++kk) {
        v16h bk = load_frag_rowmajor(&Ks[buf][nf * 16][kk * 32], 80);
        a = wmma16(aq[kk], bk, a);
      }
      s[nf] = a;
    }

    // scale + causal mask + row max
    float newmax[8];
#pragma unroll
    for (int r = 0; r < 8; ++r) newmax[r] = row_max[r];
#pragma unroll
    for (int nf = 0; nf < 8; ++nf)
#pragma unroll
      for (int r = 0; r < 8; ++r) {
        const int key = kb * 128 + nf * 16 + n_in;
        const int qr = qrow0 + r + 8 * hl;
        float v = s[nf][r] * scale;
        v = (key <= qr) ? v : NEG_INF;
        s[nf][r] = v;
        newmax[r] = fmaxf(newmax[r], v);
      }
    // reduce max across the 16-lane half that owns the row
#pragma unroll
    for (int off = 1; off < 16; off <<= 1)
#pragma unroll
      for (int r = 0; r < 8; ++r)
        newmax[r] = fmaxf(newmax[r], __shfl_xor(newmax[r], off, 32));

    float corr[8], psum[8];
#pragma unroll
    for (int r = 0; r < 8; ++r) {
      corr[r] = __expf(row_max[r] - newmax[r]);
      row_max[r] = newmax[r];
      row_sum[r] *= corr[r];
      psum[r] = 0.f;
    }
    // rescale O accumulators
#pragma unroll
    for (int j = 0; j < 4; ++j)
#pragma unroll
      for (int r = 0; r < 8; ++r) o[j][r] *= corr[r];

    // P = exp(S - max); stash to LDS in A-fragment-friendly layout
#pragma unroll
    for (int nf = 0; nf < 8; ++nf)
#pragma unroll
      for (int r = 0; r < 8; ++r) {
        const float p = __expf(s[nf][r] - newmax[r]);
        psum[r] += p;
        Ps[wid][r + 8 * hl][nf * 16 + n_in] = (_Float16)p;
      }
#pragma unroll
    for (int off = 1; off < 16; off <<= 1)
#pragma unroll
      for (int r = 0; r < 8; ++r)
        psum[r] += __shfl_xor(psum[r], off, 32);
#pragma unroll
    for (int r = 0; r < 8; ++r) row_sum[r] += psum[r];

    // O += P * V   (k = 128 keys -> 4 k-steps; n = 64 -> 4 n-frags)
#pragma unroll
    for (int kk = 0; kk < 4; ++kk) {
      v16h ap = load_frag_rowmajor(&Ps[wid][0][kk * 32], 136);
#pragma unroll
      for (int j = 0; j < 4; ++j) {
        v16h bv = load_frag_kmajor(&Vs[buf][kk * 32][j * 16], 80);
        o[j] = wmma16(ap, bv, o[j]);
      }
    }

    wait_async0();
    __syncthreads();
  }

  // normalize + store Oh[b, t, h*64 + d]
#pragma unroll
  for (int j = 0; j < 4; ++j)
#pragma unroll
    for (int r = 0; r < 8; ++r) {
      const int t = qrow0 + r + 8 * hl;
      const int d = j * 16 + n_in;
      const float val = o[j][r] / row_sum[r];
      Oh[((size_t)b * T_SZ + t) * E_SZ + h * DK_SZ + d] = (_Float16)val;
    }
}

// ---------------------------------------------------------------- launch
extern "C" void kernel_launch(void* const* d_in, const int* in_sizes, int n_in,
                              void* d_out, int out_size, void* d_ws, size_t ws_size,
                              hipStream_t stream) {
  const float* X     = (const float*)d_in[0];  // [B,T,E]
  const float* W_qkv = (const float*)d_in[1];  // [3E,E]
  const float* b_qkv = (const float*)d_in[2];  // [3E]
  const float* W_out = (const float*)d_in[3];  // [E,E]
  const float* b_out = (const float*)d_in[4];  // [E]
  float* out = (float*)d_out;

  const int M = B_SZ * T_SZ;          // 8192
  const size_t nX    = (size_t)M * E_SZ;        // 8388608
  const size_t nWqkv = (size_t)3 * E_SZ * E_SZ; // 3145728
  const size_t nWout = (size_t)E_SZ * E_SZ;     // 1048576
  const size_t nQKV  = (size_t)B_SZ * H_SZ * T_SZ * 192;

  char* ws = (char*)d_ws;
  _Float16* Xh    = (_Float16*)ws;                 ws += nX * 2;
  _Float16* Wqkvh = (_Float16*)ws;                 ws += nWqkv * 2;
  _Float16* Wouth = (_Float16*)ws;                 ws += nWout * 2;
  _Float16* qkvh  = (_Float16*)ws;                 ws += nQKV * 2;
  _Float16* Ohb   = (_Float16*)ws;

  // 1) fp32 -> fp16 conversions
  cvt_f32_f16_kernel<<<(int)(nX / 4 + 255) / 256, 256, 0, stream>>>(X, Xh, (int)(nX / 4));
  cvt_f32_f16_kernel<<<(int)(nWqkv / 4 + 255) / 256, 256, 0, stream>>>(W_qkv, Wqkvh, (int)(nWqkv / 4));
  cvt_f32_f16_kernel<<<(int)(nWout / 4 + 255) / 256, 256, 0, stream>>>(W_out, Wouth, (int)(nWout / 4));

  // 2) QKV projection: [8192,1024] x [3072,1024]^T -> scatter [B,H,T,192] f16
  gemm_f16_kernel<0><<<dim3(3 * E_SZ / 128, M / 128), 256, 0, stream>>>(
      Xh, Wqkvh, b_qkv, qkvh, nullptr, M, 3 * E_SZ, E_SZ);

  // 3) flash attention per (b,h) x 128-row query blocks
  attn_kernel<<<dim3(T_SZ / 128, B_SZ * H_SZ), 256, 0, stream>>>(qkvh, Ohb);

  // 4) output projection: [8192,1024] x [1024,1024]^T -> f32 d_out
  gemm_f16_kernel<1><<<dim3(E_SZ / 128, M / 128), 256, 0, stream>>>(
      Ohb, Wouth, b_out, nullptr, out, M, E_SZ, E_SZ);
}
